// PKMLinear_57372173140180
// MI455X (gfx1250) — compile-verified, matrix operand order checked by reading, not run on hardware
//
#include <hip/hip_runtime.h>
#include <stdint.h>

typedef __attribute__((ext_vector_type(16))) __bf16        v16bf;
typedef __attribute__((ext_vector_type(8)))  float         v8f;
typedef __attribute__((ext_vector_type(4)))  unsigned int  u32x4;
typedef __attribute__((ext_vector_type(4)))  unsigned short u16x4;

#define D_IN    2048
#define PKM     1024
#define NHEAD   8
#define BATCH   8192
#define NLAT    600000
#define KTOT    32
#define KH      4
#define BM      16
#define APITCH  2056   // halves per LDS A row (4112 B => 4-bank skew per row)
#define LPITCH2 1028   // floats per LDS logits row (half-head: 1024 cols + skew)

union FragBF { v16bf v; u32x4 q[2]; };

// ---------- f32 -> bf16 (RNE) bulk conversion, 4 elems/thread ----------
__global__ void cvt_bf16_kernel(const float* __restrict__ in,
                                unsigned short* __restrict__ out,
                                long long n4) {
  long long i = (long long)blockIdx.x * blockDim.x + threadIdx.x;
  if (i >= n4) return;
  u32x4 f = ((const u32x4*)in)[i];          // raw f32 bits
  u16x4 o;
#pragma unroll
  for (int j = 0; j < 4; ++j) {
    unsigned u = f[j];
    unsigned r = u + 0x7FFFu + ((u >> 16) & 1u);
    o[j] = (unsigned short)(r >> 16);
  }
  ((u16x4*)out)[i] = o;
}

// ---------- wave32 top-4 over 1024 LDS floats (stable, min-index ties) ----------
__device__ inline void wave_topk4(const float* __restrict__ base, int lane,
                                  float* wv, int* iv) {
  int sel0 = -1, sel1 = -1, sel2 = -1;
#pragma unroll
  for (int t = 0; t < KH; ++t) {
    float best = -3.402823466e38f;
    int bidx = 1 << 30;
    for (int c = lane; c < PKM; c += 32) {
      if (c == sel0 || c == sel1 || c == sel2) continue;
      float v = base[c];
      if (v > best || (v == best && c < bidx)) { best = v; bidx = c; }
    }
#pragma unroll
    for (int off = 16; off > 0; off >>= 1) {
      float ov = __shfl_xor(best, off, 32);
      int   oi = __shfl_xor(bidx, off, 32);
      if (ov > best || (ov == best && oi < bidx)) { best = ov; bidx = oi; }
    }
    wv[t] = best; iv[t] = bidx;
    if (t == 0) sel0 = bidx; else if (t == 1) sel1 = bidx; else if (t == 2) sel2 = bidx;
  }
}

// ---------- one half-head GEMM (16 rows x 1024 cols), logits -> LDS ----------
// Round-2 style inner loop: fragments loaded in-loop (no manual rotation),
// which the backend schedules into load clauses without spilling.
__device__ inline void gemm_half(const unsigned short* __restrict__ arow,  // +laneN*APITCH
                                 const unsigned short* __restrict__ Wb,
                                 const float* __restrict__ bias,
                                 int colbase,                               // hbase + half*PKM
                                 float* __restrict__ ldsL,
                                 int wave, int laneN, int laneHi, int koff) {
#pragma unroll 1
  for (int pass = 0; pass < PKM / 256; ++pass) {
    const int nloc0 = pass * 256 + wave * 32 + laneN;   // 0..1023 within half
    const int nloc1 = nloc0 + 16;
    const unsigned short* wrow0 = Wb + (size_t)(colbase + nloc0) * D_IN;
    const unsigned short* wrow1 = Wb + (size_t)(colbase + nloc1) * D_IN;
    v8f acc0 = {};
    v8f acc1 = {};
    for (int k0 = 0; k0 < D_IN; k0 += 32) {
      FragBF a, b0, b1;
      a.q[0]  = *(const u32x4*)(arow  + k0 + koff);       // K = k0+koff .. +7
      a.q[1]  = *(const u32x4*)(arow  + k0 + koff + 16);  // K = k0+koff+16 .. +23
      b0.q[0] = *(const u32x4*)(wrow0 + k0 + koff);
      b0.q[1] = *(const u32x4*)(wrow0 + k0 + koff + 16);
      b1.q[0] = *(const u32x4*)(wrow1 + k0 + koff);
      b1.q[1] = *(const u32x4*)(wrow1 + k0 + koff + 16);
      acc0 = __builtin_amdgcn_wmma_f32_16x16x32_bf16(
          false, a.v, false, b0.v, (short)0, acc0, false, false);
      acc1 = __builtin_amdgcn_wmma_f32_16x16x32_bf16(
          false, a.v, false, b1.v, (short)0, acc1, false, false);
    }
    const float bv0 = bias[colbase + nloc0];
    const float bv1 = bias[colbase + nloc1];
#pragma unroll
    for (int j = 0; j < 8; ++j) {                        // D: M = j + laneHi*8
      ldsL[(j + laneHi * 8) * LPITCH2 + nloc0] = acc0[j] + bv0;
      ldsL[(j + laneHi * 8) * LPITCH2 + nloc1] = acc1[j] + bv1;
    }
  }
}

// ---------- fused bf16-WMMA GEMM + per-head PKM top-k ----------
// grid = (BATCH/BM, NHEAD), block = 256 (8 wave32)
// Head processed as two independent 1024-col halves through one 64KB LDS
// logits buffer => ~128.5 KB LDS total => 2 blocks/WGP (4 waves/SIMD).
__global__ __launch_bounds__(256) void pkm_fused(
    const unsigned short* __restrict__ xb,   // bf16 bits, (BATCH, D_IN)
    const unsigned short* __restrict__ Wb,   // bf16 bits, (NHEAD*2*PKM, D_IN)
    const float* __restrict__ bias,          // (NHEAD*2*PKM)
    float* __restrict__ wmid,                // (BATCH, NHEAD, KH)
    int* __restrict__ imid) {
  extern __shared__ char smem[];
  unsigned short* ldsA = (unsigned short*)smem;                         // 16*2056*2 B
  float* ldsL = (float*)(smem + (size_t)BM * APITCH * sizeof(unsigned short));

  const int row0  = blockIdx.x * BM;
  const int head  = blockIdx.y;
  const int hbase = head * (2 * PKM);

  // ---- stage x tile (16 x 2048 bf16) into LDS ----
  for (int t = threadIdx.x; t < BM * (D_IN / 8); t += 256) {
    int r  = t / (D_IN / 8);
    int kc = (t % (D_IN / 8)) * 8;
    u32x4 v = *(const u32x4*)(xb + (size_t)(row0 + r) * D_IN + kc);
    *(u32x4*)(ldsA + r * APITCH + kc) = v;
  }
  __syncthreads();

  const int wave   = threadIdx.x >> 5;
  const int lane   = threadIdx.x & 31;
  const int laneN  = lane & 15;
  const int laneHi = lane >> 4;          // 0: low half-wave, 1: high half-wave
  const int koff   = laneHi ? 8 : 0;     // ISA 16-bit A/B lane K-group offset

  const unsigned short* arow = ldsA + laneN * APITCH;  // A: M = laneN

  float w1[2][KH], w2[2][KH];
  int   i1[2][KH], i2[2][KH];

  // ---- half 1 (x1: cols 0..1023 of head) ----
  gemm_half(arow, Wb, bias, hbase, ldsL, wave, laneN, laneHi, koff);
  __syncthreads();
  for (int rr = 0; rr < 2; ++rr) {
    const int row = wave + rr * 8;
    wave_topk4(ldsL + row * LPITCH2, lane, w1[rr], i1[rr]);
  }
  __syncthreads();

  // ---- half 2 (x2: cols 1024..2047 of head) ----
  gemm_half(arow, Wb, bias, hbase + PKM, ldsL, wave, laneN, laneHi, koff);
  __syncthreads();
  for (int rr = 0; rr < 2; ++rr) {
    const int row = wave + rr * 8;
    wave_topk4(ldsL + row * LPITCH2, lane, w2[rr], i2[rr]);
  }

  // ---- combine: 4x4 sums, relu, mask, stable top-4 of 16 ----
  for (int rr = 0; rr < 2; ++rr) {
    const int row = wave + rr * 8;
    float cw[16]; int ci[16];
#pragma unroll
    for (int a = 0; a < 4; ++a)
#pragma unroll
      for (int bq = 0; bq < 4; ++bq) {
        int p  = a * 4 + bq;
        float v = w1[rr][a] + w2[rr][bq];
        v = v > 0.f ? v : 0.f;                 // relu
        int gi = i1[rr][a] * PKM + i2[rr][bq];
        if (gi >= NLAT) v = -1.0f;             // mask
        cw[p] = v; ci[p] = gi;
      }
    if (lane == 0) {
      for (int t = 0; t < KH; ++t) {           // stable top-4 of 16
        float best = -3.402823466e38f; int bp = 0;
        for (int p = 0; p < 16; ++p)
          if (cw[p] > best) { best = cw[p]; bp = p; }
        cw[bp] = -3.402823466e38f;
        int gi = ci[bp];
        float wf = (gi < NLAT) ? best : 0.0f;  // w *= (i < L)
        int gout = (((row0 + row) * NHEAD) + head) * KH + t;
        wmid[gout] = wf;
        imid[gout] = gi < NLAT ? gi : (NLAT - 1);  // min(i, L-1)
      }
    }
  }
}

// ---------- final stable 32-wide descending sort per row ----------
__global__ void pkm_final(const float* __restrict__ wmid,
                          const int* __restrict__ imid,
                          float* __restrict__ outw, int* __restrict__ outi) {
  int row = blockIdx.x * blockDim.x + threadIdx.x;
  if (row >= BATCH) return;
  float w[KTOT]; int id[KTOT]; bool used[KTOT];
#pragma unroll
  for (int t = 0; t < KTOT; ++t) {
    w[t] = wmid[(size_t)row * KTOT + t];
    id[t] = imid[(size_t)row * KTOT + t];
    used[t] = false;
  }
  for (int t = 0; t < KTOT; ++t) {
    float best = -3.402823466e38f; int bp = 0;
    for (int p = 0; p < KTOT; ++p)
      if (!used[p] && w[p] > best) { best = w[p]; bp = p; }
    used[bp] = true;
    outw[(size_t)row * KTOT + t] = best;
    outi[(size_t)row * KTOT + t] = id[bp];
  }
}

extern "C" void kernel_launch(void* const* d_in, const int* in_sizes, int n_in,
                              void* d_out, int out_size, void* d_ws, size_t ws_size,
                              hipStream_t stream) {
  const float* x = (const float*)d_in[0];   // (8192, 2048)
  const float* W = (const float*)d_in[1];   // (16384, 2048)
  const float* b = (const float*)d_in[2];   // (16384,)
  // d_in[3] = k (compile-time 32)

  char* ws = (char*)d_ws;
  const size_t xb_bytes = (size_t)BATCH * D_IN * 2;                 // 32 MB
  const size_t wb_bytes = (size_t)NHEAD * 2 * PKM * D_IN * 2;       // 64 MB
  unsigned short* xb   = (unsigned short*)ws;
  unsigned short* Wb   = (unsigned short*)(ws + xb_bytes);
  float*          wmid = (float*)(ws + xb_bytes + wb_bytes);        // 1 MB
  int*            imid = (int*)((char*)wmid + (size_t)BATCH * KTOT * 4);

  long long n4x = (long long)BATCH * D_IN / 4;
  cvt_bf16_kernel<<<(unsigned)((n4x + 255) / 256), 256, 0, stream>>>(x, xb, n4x);
  long long n4w = (long long)NHEAD * 2 * PKM * D_IN / 4;
  cvt_bf16_kernel<<<(unsigned)((n4w + 255) / 256), 256, 0, stream>>>(W, Wb, n4w);

  size_t shmem = (size_t)BM * APITCH * 2 + (size_t)BM * LPITCH2 * 4; // ~128.5 KB
  dim3 grid(BATCH / BM, NHEAD);
  pkm_fused<<<grid, 256, shmem, stream>>>(xb, Wb, b, wmid, imid);

  pkm_final<<<(BATCH + 255) / 256, 256, 0, stream>>>(
      wmid, imid, (float*)d_out, ((int*)d_out) + (size_t)BATCH * KTOT);
}
